// SAGECreditRisk_67680094650381
// MI455X (gfx1250) — matrix operand order repaired and checked by reading, not compiled, not run on hardware
//
#include <hip/hip_runtime.h>
#include <hip/hip_bf16.h>

// GraphSAGE (mean aggr) 2-layer + linear head, N=100k, E=1.6M, D=H=128.
// All intermediates are L2-resident (192MB L2); GEMMs run on v_wmma_f32_16x16x32_f16.

typedef __attribute__((ext_vector_type(16))) _Float16 v16h;
typedef __attribute__((ext_vector_type(8)))  float    v8f;

// ---------------------------------------------------------------- zero ----
__global__ __launch_bounds__(256) void k_zero(float* __restrict__ p, long n) {
    long i      = (long)blockIdx.x * blockDim.x + threadIdx.x;
    long stride = (long)gridDim.x * blockDim.x;
    for (; i < n; i += stride) p[i] = 0.0f;
}

// ------------------------------------------------------------- degrees ----
__global__ __launch_bounds__(256) void k_count(const int* __restrict__ dst, int E,
                                               float* __restrict__ cnt) {
    int e = blockIdx.x * blockDim.x + threadIdx.x;
    if (e < E) unsafeAtomicAdd(&cnt[dst[e]], 1.0f);
}

// ------------------------------------------------- edge scatter (sum) -----
// One edge per wave32: src/dst are wave-uniform (scalar loads); each lane
// moves 4 consecutive features: 16B vector load + 4 f32 L2 atomics.
__global__ __launch_bounds__(256) void k_scatter(const float* __restrict__ h,
                                                 const int* __restrict__ src,
                                                 const int* __restrict__ dst,
                                                 float* __restrict__ agg, int E) {
    long tid = (long)blockIdx.x * blockDim.x + threadIdx.x;
    int  e   = (int)(tid >> 5);
    if (e >= E) return;
    int c = (int)(tid & 31) * 4;
    int s = src[e];
    int d = dst[e];
    if (e + 4096 < E) {               // stream the edge list ahead of us
        __builtin_prefetch(&src[e + 4096], 0, 0);
        __builtin_prefetch(&dst[e + 4096], 0, 0);
    }
    float4 v = *(const float4*)(h + (long)s * 128 + c);
    float* a = agg + (long)d * 128 + c;
    unsafeAtomicAdd(a + 0, v.x);
    unsafeAtomicAdd(a + 1, v.y);
    unsafeAtomicAdd(a + 2, v.z);
    unsafeAtomicAdd(a + 3, v.w);
}

// ----------------------------------------------------- fused SAGE layer ---
// h_out = relu( (agg/deg) @ Wn + h_in @ Wr + b )      (FUSE==false)
// out   = relu(...) @ Wh + bh                          (FUSE==true, head fused)
//
// Block = 256 threads = 8 waves; each wave owns one 16-row tile (block: 128 rows).
// Weights staged to LDS as f16, TRANSPOSED (wsh[n][k] = W[k][n]) so a lane's
// WMMA B-fragment (column N = lane%16, K = chunk*32 + (lane/16)*16 .. +16)
// is 32 contiguous bytes in LDS.
template <bool FUSE>
__global__ __launch_bounds__(256) void k_sage(const float* __restrict__ agg,
                                              const float* __restrict__ hin,
                                              const float* __restrict__ Wn,
                                              const float* __restrict__ Wr,
                                              const float* __restrict__ bias,
                                              const float* __restrict__ cnt,
                                              float* __restrict__ outp,
                                              const float* __restrict__ Wh,
                                              const float* __restrict__ bhp,
                                              int N) {
    __shared__ _Float16 wn_s[128 * 128];   // 32 KB
    __shared__ _Float16 wr_s[128 * 128];   // 32 KB

    const int tid = threadIdx.x;
    // Stage + transpose + f16-convert both weight matrices (coalesced reads).
    for (int i = tid; i < 128 * 128; i += 256) {
        int k = i >> 7, n = i & 127;
        wn_s[n * 128 + k] = (_Float16)Wn[i];
        wr_s[n * 128 + k] = (_Float16)Wr[i];
    }
    __syncthreads();

    const int wave = tid >> 5;
    const int lane = tid & 31;
    const int tile = blockIdx.x * 8 + wave;
    if (tile * 16 >= N) return;            // wave-uniform: EXEC stays all-ones

    const int half_sel = lane >> 4;        // 0: K-lo half, 1: K-hi half
    const int lidx     = lane & 15;        // row (A) / column (B,C,D) id
    const int row      = tile * 16 + lidx;

    const float inv = 1.0f / fmaxf(cnt[row], 1.0f);   // mean = sum / max(deg,1)

    // ---- A fragments, contiguous k order (same as B):
    //   lane holds row M=lidx, halfs K = base..base+16,
    //   base = half_sel*16, per 32-wide K chunk.
    const float* __restrict__ arow = agg + (long)row * 128;
    const float* __restrict__ hrow = hin + (long)row * 128;
    v16h Aa[4], Ah[4];
#pragma unroll
    for (int c = 0; c < 4; ++c) {
        const int base = c * 32 + half_sel * 16;
        float4 a0 = *(const float4*)(arow + base);
        float4 a1 = *(const float4*)(arow + base + 4);
        float4 a2 = *(const float4*)(arow + base + 8);
        float4 a3 = *(const float4*)(arow + base + 12);
        float4 h0 = *(const float4*)(hrow + base);
        float4 h1 = *(const float4*)(hrow + base + 4);
        float4 h2 = *(const float4*)(hrow + base + 8);
        float4 h3 = *(const float4*)(hrow + base + 12);
        Aa[c][0]  = (_Float16)(a0.x * inv);  Aa[c][1]  = (_Float16)(a0.y * inv);
        Aa[c][2]  = (_Float16)(a0.z * inv);  Aa[c][3]  = (_Float16)(a0.w * inv);
        Aa[c][4]  = (_Float16)(a1.x * inv);  Aa[c][5]  = (_Float16)(a1.y * inv);
        Aa[c][6]  = (_Float16)(a1.z * inv);  Aa[c][7]  = (_Float16)(a1.w * inv);
        Aa[c][8]  = (_Float16)(a2.x * inv);  Aa[c][9]  = (_Float16)(a2.y * inv);
        Aa[c][10] = (_Float16)(a2.z * inv);  Aa[c][11] = (_Float16)(a2.w * inv);
        Aa[c][12] = (_Float16)(a3.x * inv);  Aa[c][13] = (_Float16)(a3.y * inv);
        Aa[c][14] = (_Float16)(a3.z * inv);  Aa[c][15] = (_Float16)(a3.w * inv);
        Ah[c][0]  = (_Float16)h0.x;  Ah[c][1]  = (_Float16)h0.y;
        Ah[c][2]  = (_Float16)h0.z;  Ah[c][3]  = (_Float16)h0.w;
        Ah[c][4]  = (_Float16)h1.x;  Ah[c][5]  = (_Float16)h1.y;
        Ah[c][6]  = (_Float16)h1.z;  Ah[c][7]  = (_Float16)h1.w;
        Ah[c][8]  = (_Float16)h2.x;  Ah[c][9]  = (_Float16)h2.y;
        Ah[c][10] = (_Float16)h2.z;  Ah[c][11] = (_Float16)h2.w;
        Ah[c][12] = (_Float16)h3.x;  Ah[c][13] = (_Float16)h3.y;
        Ah[c][14] = (_Float16)h3.z;  Ah[c][15] = (_Float16)h3.w;
    }

    float logit[8];
#pragma unroll
    for (int r = 0; r < 8; ++r) logit[r] = 0.0f;

    // ---- 8 column tiles of 16; 4 K chunks x 2 matrices = 8 WMMAs each.
#pragma unroll
    for (int n = 0; n < 8; ++n) {
        const int col = n * 16 + lidx;
        const float bv = bias[col];        // bias folded into accumulator
        v8f acc;
#pragma unroll
        for (int r = 0; r < 8; ++r) acc[r] = bv;

#pragma unroll
        for (int c = 0; c < 4; ++c) {
            const v16h bn = *(const v16h*)(wn_s + col * 128 + c * 32 + half_sel * 16);
            acc = __builtin_amdgcn_wmma_f32_16x16x32_f16(
                false, Aa[c], false, bn, (short)0, acc, false, false);
            const v16h br = *(const v16h*)(wr_s + col * 128 + c * 32 + half_sel * 16);
            acc = __builtin_amdgcn_wmma_f32_16x16x32_f16(
                false, Ah[c], false, br, (short)0, acc, false, false);
        }

        if (!FUSE) {
            // C/D layout: element r -> row M = half_sel*8 + r, column = col
#pragma unroll
            for (int r = 0; r < 8; ++r)
                outp[(long)(tile * 16 + half_sel * 8 + r) * 128 + col] =
                    fmaxf(acc[r], 0.0f);
        } else {
            const float whv = Wh[col];
#pragma unroll
            for (int r = 0; r < 8; ++r)
                logit[r] += fmaxf(acc[r], 0.0f) * whv;
        }
    }

    if (FUSE) {
        // Reduce over the 16 columns held by this half-wave (xor stays in-half).
#pragma unroll
        for (int off = 8; off >= 1; off >>= 1)
#pragma unroll
            for (int r = 0; r < 8; ++r)
                logit[r] += __shfl_xor(logit[r], off, 32);
        if (lidx == 0) {
            const float bh = bhp[0];
#pragma unroll
            for (int r = 0; r < 8; ++r)
                outp[tile * 16 + half_sel * 8 + r] = logit[r] + bh;
        }
    }
}

// ---------------------------------------------------------------- launch --
extern "C" void kernel_launch(void* const* d_in, const int* in_sizes, int n_in,
                              void* d_out, int out_size, void* d_ws, size_t ws_size,
                              hipStream_t stream) {
    const float* x   = (const float*)d_in[0];
    const int*   ei  = (const int*)  d_in[1];
    const float* Wn0 = (const float*)d_in[2];
    const float* Wr0 = (const float*)d_in[3];
    const float* b0  = (const float*)d_in[4];
    const float* Wn1 = (const float*)d_in[5];
    const float* Wr1 = (const float*)d_in[6];
    const float* b1  = (const float*)d_in[7];
    const float* Wh  = (const float*)d_in[8];
    const float* bh  = (const float*)d_in[9];

    const int N = in_sizes[0] / 128;
    const int E = in_sizes[1] / 2;
    const int* src = ei;         // edge_index[0]
    const int* dst = ei + E;     // edge_index[1]

    float* ws  = (float*)d_ws;
    float* cnt = ws;                          // N floats (N % 4 == 0 -> aligned)
    float* agg = ws + N;                      // N*128 floats
    float* h0  = ws + N + (long)N * 128;      // N*128 floats
    float* out = (float*)d_out;               // N logits

    const long aggN = (long)N * 128;

    // 1) zero cnt + agg
    k_zero<<<2048, 256, 0, stream>>>(cnt, (long)N + aggN);
    // 2) degrees
    k_count<<<(E + 255) / 256, 256, 0, stream>>>(dst, E, cnt);
    // 3) scatter x -> agg
    {
        long thr = (long)E * 32;
        k_scatter<<<(int)((thr + 255) / 256), 256, 0, stream>>>(x, src, dst, agg, E);
    }
    // 4) layer 0 -> h0
    {
        int tiles  = (N + 15) / 16;
        int blocks = (tiles + 7) / 8;
        k_sage<false><<<blocks, 256, 0, stream>>>(agg, x, Wn0, Wr0, b0, cnt, h0,
                                                  nullptr, nullptr, N);
    }
    // 5) re-zero agg, scatter h0 -> agg
    k_zero<<<2048, 256, 0, stream>>>(agg, aggN);
    {
        long thr = (long)E * 32;
        k_scatter<<<(int)((thr + 255) / 256), 256, 0, stream>>>(h0, src, dst, agg, E);
    }
    // 6) layer 1 + fused head -> logits
    {
        int tiles  = (N + 15) / 16;
        int blocks = (tiles + 7) / 8;
        k_sage<true><<<blocks, 256, 0, stream>>>(agg, h0, Wn1, Wr1, b1, cnt, out,
                                                 Wh, bh, N);
    }
}
